// LinearAttention_87557203296527
// MI455X (gfx1250) — matrix operand order repaired and checked by reading, not compiled, not run on hardware
//
#include <hip/hip_runtime.h>
#include <math.h>
#include <stdint.h>

// ---------------------------------------------------------------------------
// LinearAttention on MI455X (gfx1250), wave32.
// GEMMs use bf16-split WMMA (3xBF16 ~ fp32 accuracy, 3/8 the cost of the
// fp32 16x16x4 WMMA path). Panel GEMMs stage the shared activation panel in
// LDS via GLOBAL_LOAD_ASYNC_TO_LDS_B32 (ASYNCcnt), double-buffered so the
// async fill of step k+1 overlaps the 12 WMMAs of step k.
// ---------------------------------------------------------------------------

typedef __attribute__((ext_vector_type(16))) __bf16 v16bf;
typedef __attribute__((ext_vector_type(8)))  float  v8f;

#define BATCH 8
#define CIN   256
#define NPIX  4096   // 64*64
#define HID   512    // 4 heads * 128
#define QKVO  1536   // 3*HID

#if defined(__gfx1250__)
#define HAVE_ASYNC_LDS 1
#else
#define HAVE_ASYNC_LDS 0
#endif

// --- split-bf16 multiply-accumulate: acc += A*B with ~fp32 accuracy ---------
__device__ __forceinline__ v8f wmma3(v16bf ah, v16bf al, v16bf bh, v16bf bl, v8f c) {
    c = __builtin_amdgcn_wmma_f32_16x16x32_bf16(false, ah, false, bh, (short)0, c, false, false);
    c = __builtin_amdgcn_wmma_f32_16x16x32_bf16(false, al, false, bh, (short)0, c, false, false);
    c = __builtin_amdgcn_wmma_f32_16x16x32_bf16(false, ah, false, bl, (short)0, c, false, false);
    return c;
}

__device__ __forceinline__ void split_bf16(float x, __bf16& hi, __bf16& lo) {
    hi = (__bf16)x;
    lo = (__bf16)(x - (float)hi);
}

// A tile 16(M)x32(K), bf16 layout: lane l -> M = l&15; slot s=2v+p ->
// K = 2v + p + (l>=16 ? 16 : 0). A row-major [M][K].
__device__ __forceinline__ void load_a(const float* __restrict__ A, int lda,
                                       int m0, int k0, int lane,
                                       v16bf& ah, v16bf& al) {
    const float* row = A + (size_t)(m0 + (lane & 15)) * lda + k0 + ((lane & 16) ? 16 : 0);
    #pragma unroll
    for (int v = 0; v < 8; ++v) {
        int kb = 2 * v;
        #pragma unroll
        for (int p = 0; p < 2; ++p) {
            __bf16 hi, lo; split_bf16(row[kb + p], hi, lo);
            ah[2 * v + p] = hi; al[2 * v + p] = lo;
        }
    }
}

// A tile where the source is stored transposed: A[M][K] = At[K][M].
__device__ __forceinline__ void load_a_tr(const float* __restrict__ At, int ldat,
                                          int m0, int k0, int lane,
                                          v16bf& ah, v16bf& al) {
    int m    = m0 + (lane & 15);
    int koff = k0 + ((lane & 16) ? 16 : 0);
    #pragma unroll
    for (int v = 0; v < 8; ++v) {
        int kb = 2 * v;
        #pragma unroll
        for (int p = 0; p < 2; ++p) {
            __bf16 hi, lo; split_bf16(At[(size_t)(koff + kb + p) * ldat + m], hi, lo);
            ah[2 * v + p] = hi; al[2 * v + p] = lo;
        }
    }
}

// B tile 32(K)x16(N), bf16 layout: lane l -> N = l&15; slot s -> K = s + (l>=16?16:0).
// B row-major [K][N] in global memory.
__device__ __forceinline__ void load_b(const float* __restrict__ B, int ldb,
                                       int k0, int n0, int lane,
                                       v16bf& bh, v16bf& bl) {
    int n     = n0 + (lane & 15);
    int kbase = k0 + ((lane & 16) ? 16 : 0);
    #pragma unroll
    for (int s = 0; s < 16; ++s) {
        __bf16 hi, lo; split_bf16(B[(size_t)(kbase + s) * ldb + n], hi, lo);
        bh[s] = hi; bl[s] = lo;
    }
}

// B tile where source stored transposed: B[K][N] = Bt[N][K] (contiguous K per lane).
__device__ __forceinline__ void load_b_tr(const float* __restrict__ Bt, int ldbt,
                                          int k0, int n0, int lane,
                                          v16bf& bh, v16bf& bl) {
    const float* p = Bt + (size_t)(n0 + (lane & 15)) * ldbt + k0 + ((lane & 16) ? 16 : 0);
    #pragma unroll
    for (int s = 0; s < 16; ++s) {
        __bf16 hi, lo; split_bf16(p[s], hi, lo);
        bh[s] = hi; bl[s] = lo;
    }
}

// B tile read from an LDS panel ldsB[32][LDSW] (K x N, padded rows).
#define LDSW 65
__device__ __forceinline__ void load_b_lds(const float* ldsB, int ncol, int lane,
                                           v16bf& bh, v16bf& bl) {
    const float* p = ldsB + ((lane & 16) ? 16 * LDSW : 0) + ncol + (lane & 15);
    #pragma unroll
    for (int s = 0; s < 16; ++s) {
        __bf16 hi, lo; split_bf16(p[s * LDSW], hi, lo);
        bh[s] = hi; bl[s] = lo;
    }
}

// C/D 16x16 f32: component i -> M = i + (l>=16?8:0); N = l&15. Row-major store.
__device__ __forceinline__ void store_c(float* __restrict__ C, int ldc,
                                        int m0, int n0, int lane, v8f acc,
                                        const float* __restrict__ bias) {
    int n  = n0 + (lane & 15);
    int mo = m0 + ((lane & 16) ? 8 : 0);
    #pragma unroll
    for (int i = 0; i < 8; ++i) {
        float v = acc[i];
        if (bias) v += bias[mo + i];
        C[(size_t)(mo + i) * ldc + n] = v;
    }
}

// --- async copy of one dword into LDS (CDNA5 GLOBAL_LOAD_ASYNC_TO_LDS_B32) --
__device__ __forceinline__ void async_b32(void* lds_dst, const float* g) {
#if HAVE_ASYNC_LDS
    uint32_t lds_off = (uint32_t)(uintptr_t)lds_dst;   // generic->LDS = low 32 bits
    asm volatile("global_load_async_to_lds_b32 %0, %1, off"
                 :: "v"(lds_off), "v"((unsigned long long)(uintptr_t)g)
                 : "memory");
#else
    *(float*)lds_dst = *g;
#endif
}
__device__ __forceinline__ void async_drain() {
#if HAVE_ASYNC_LDS
    asm volatile("s_wait_asynccnt 0" ::: "memory");
#endif
}

// ---------------------------------------------------------------------------
// Panel GEMM: C[b, m, n] = sum_k A[m,k] * Bp[b, k, n] (+ bias[m])
//   A       : [Mtot][KTOT] weights (L2-resident)
//   Bp      : activation panel, ldb = NPIX, per-batch stride bstrB
//   block   : 256 thr = 8 waves, block tile 128(M) x 64(N), wave tile 32x32
//   LDS     : B panel 32(K) x 64(N), double-buffered async-to-LDS pipeline
// grid: (NPIX/64, Mtot/128, BATCH)
// ---------------------------------------------------------------------------
template <int KTOT, bool BIAS>
__global__ void panel_gemm(const float* __restrict__ A,
                           const float* __restrict__ Bp, size_t bstrB,
                           float* __restrict__ Cp, size_t bstrC,
                           const float* __restrict__ bias) {
    __shared__ float ldsB[2][32 * LDSW];
    const int tid = threadIdx.x, lane = tid & 31, wave = tid >> 5;
    const int nblk = blockIdx.x * 64;
    const int mblk = blockIdx.y * 128;
    const int b    = blockIdx.z;
    const int wm = mblk + (wave >> 1) * 32;       // wave M origin
    const int wn = (wave & 1) * 32;               // wave N origin within panel
    const float* Bb = Bp + (size_t)b * bstrB;
    float*       Cb = Cp + (size_t)b * bstrC;

    // cooperative copy pattern: thread -> row kr, 8 consecutive columns
    const int kr = tid >> 3;                      // 0..31
    const int nc = (tid & 7) * 8;                 // 0..56
    const float* gk = Bb + (size_t)kr * NPIX + nblk + nc;   // k0 added per step

    // prologue: fill buffer 0
    {
        float* ldst = &ldsB[0][kr * LDSW + nc];
        #pragma unroll
        for (int j = 0; j < 8; ++j) async_b32(ldst + j, gk + j);
        async_drain();
    }
    __syncthreads();

    v8f a00 = {}, a01 = {}, a10 = {}, a11 = {};
    int buf = 0;
    for (int k0 = 0; k0 < KTOT; k0 += 32) {
        // issue async fill of the NEXT panel into the idle buffer
        if (k0 + 32 < KTOT) {
            const float* gsrc = gk + (size_t)(k0 + 32) * NPIX;
            float* ldst = &ldsB[buf ^ 1][kr * LDSW + nc];
            #pragma unroll
            for (int j = 0; j < 8; ++j) async_b32(ldst + j, gsrc + j);
            __builtin_prefetch(A + (size_t)(wm + (lane & 15)) * KTOT + k0 + 32, 0, 1);
        }

        // compute step k from the current buffer (overlaps the async fill)
        v16bf a0h, a0l, a1h, a1l, b0h, b0l, b1h, b1l;
        load_a(A, KTOT, wm,      k0, lane, a0h, a0l);
        load_a(A, KTOT, wm + 16, k0, lane, a1h, a1l);
        load_b_lds(ldsB[buf], wn,      lane, b0h, b0l);
        load_b_lds(ldsB[buf], wn + 16, lane, b1h, b1l);
        a00 = wmma3(a0h, a0l, b0h, b0l, a00);
        a01 = wmma3(a0h, a0l, b1h, b1l, a01);
        a10 = wmma3(a1h, a1l, b0h, b0l, a10);
        a11 = wmma3(a1h, a1l, b1h, b1l, a11);

        async_drain();        // next panel landed in LDS
        __syncthreads();      // all waves done reading buf, next buf visible
        buf ^= 1;
    }
    const float* bp = BIAS ? bias : nullptr;
    store_c(Cb, NPIX, wm,      nblk + wn,      lane, a00, bp);
    store_c(Cb, NPIX, wm,      nblk + wn + 16, lane, a01, bp);
    store_c(Cb, NPIX, wm + 16, nblk + wn,      lane, a10, bp);
    store_c(Cb, NPIX, wm + 16, nblk + wn + 16, lane, a11, bp);
}

// ---------------------------------------------------------------------------
// Softmax over n for the K region (rows o in [512,1024) per batch).
// one block of 256 threads per row; 16 elements per thread in registers
// ---------------------------------------------------------------------------
__global__ void softmax_k(float* __restrict__ qkv) {
    const int r = blockIdx.x;           // 0..4095
    const int b = r >> 9;
    const int o = HID + (r & 511);
    float* row = qkv + (size_t)b * QKVO * NPIX + (size_t)o * NPIX;
    const int tid = threadIdx.x, lane = tid & 31, wave = tid >> 5;
    __shared__ float red[8];
    float vals[16];
    float mx = -INFINITY;
    #pragma unroll
    for (int j = 0; j < 16; ++j) { vals[j] = row[tid + j * 256]; mx = fmaxf(mx, vals[j]); }
    #pragma unroll
    for (int off = 16; off > 0; off >>= 1) mx = fmaxf(mx, __shfl_xor(mx, off, 32));
    if (lane == 0) red[wave] = mx;
    __syncthreads();
    float bmx = red[0];
    #pragma unroll
    for (int i = 1; i < 8; ++i) bmx = fmaxf(bmx, red[i]);
    float s = 0.f;
    #pragma unroll
    for (int j = 0; j < 16; ++j) { vals[j] = __expf(vals[j] - bmx); s += vals[j]; }
    #pragma unroll
    for (int off = 16; off > 0; off >>= 1) s += __shfl_xor(s, off, 32);
    __syncthreads();
    if (lane == 0) red[wave] = s;
    __syncthreads();
    float bs = 0.f;
    #pragma unroll
    for (int i = 0; i < 8; ++i) bs += red[i];
    const float inv = 1.f / bs;
    #pragma unroll
    for (int j = 0; j < 16; ++j) row[tid + j * 256] = vals[j] * inv;
}

// ---------------------------------------------------------------------------
// context[bh, d, e] = sum_n k_sm[d,n] * v[e,n]   (M=128, N=128, K=4096)
// grid (8, 8, 32), one wave per block; per-lane loads fully contiguous
// ---------------------------------------------------------------------------
__global__ void context_gemm(const float* __restrict__ qkv, float* __restrict__ ctx) {
    const int lane = threadIdx.x & 31;
    const int bh = blockIdx.z, b = bh >> 2, h = bh & 3;
    const float* kk = qkv + (size_t)b * QKVO * NPIX + (size_t)(HID + h * 128) * NPIX;
    const float* vv = qkv + (size_t)b * QKVO * NPIX + (size_t)(2 * HID + h * 128) * NPIX;
    const int m0 = blockIdx.y * 16, n0 = blockIdx.x * 16;
    v8f acc = {};
    for (int k0 = 0; k0 < NPIX; k0 += 32) {
        v16bf ah, al, bhv, blv;
        load_a(kk, NPIX, m0, k0, lane, ah, al);        // A[d][n]
        load_b_tr(vv, NPIX, k0, n0, lane, bhv, blv);   // B[n][e] = v[e][n]
        acc = wmma3(ah, al, bhv, blv, acc);
    }
    store_c(ctx + (size_t)bh * 128 * 128, 128, m0, n0, lane, acc, nullptr);
}

// ---------------------------------------------------------------------------
// attn[bh, e, n] = sum_d ctx[d,e] * q[d,n]   (M=128, N=4096, K=128)
// Writes over the (now dead) V region of the qkv scratch.
// grid (256, 8, 32), one wave per block
// ---------------------------------------------------------------------------
__global__ void attn_out_gemm(float* __restrict__ qkv, const float* __restrict__ ctx) {
    const int lane = threadIdx.x & 31;
    const int bh = blockIdx.z, b = bh >> 2, h = bh & 3;
    const float* q = qkv + (size_t)b * QKVO * NPIX + (size_t)(h * 128) * NPIX;
    const float* c = ctx + (size_t)bh * 128 * 128;
    float* outp = qkv + (size_t)b * QKVO * NPIX + (size_t)(2 * HID + h * 128) * NPIX;
    const int m0 = blockIdx.y * 16, n0 = blockIdx.x * 16;
    v8f acc = {};
    for (int k0 = 0; k0 < 128; k0 += 32) {
        v16bf ah, al, bhv, blv;
        load_a_tr(c, 128, m0, k0, lane, ah, al);   // A[e][d] = ctx[d][e]
        load_b(q, NPIX, k0, n0, lane, bhv, blv);   // B[d][n] = q[d][n]
        acc = wmma3(ah, al, bhv, blv, acc);
    }
    store_c(outp, NPIX, m0, n0, lane, acc, nullptr);
}

// ---------------------------------------------------------------------------
extern "C" void kernel_launch(void* const* d_in, const int* in_sizes, int n_in,
                              void* d_out, int out_size, void* d_ws, size_t ws_size,
                              hipStream_t stream) {
    const float* x     = (const float*)d_in[0];
    const float* w_qkv = (const float*)d_in[1];
    const float* w_out = (const float*)d_in[2];
    const float* b_out = (const float*)d_in[3];
    float* y = (float*)d_out;

    // workspace: qkv (8*1536*4096 f32 = 201.3 MB) | ctx (32*128*128 f32 = 2 MB)
    float* qkv = (float*)d_ws;
    float* ctx = qkv + (size_t)BATCH * QKVO * NPIX;
    const float* attn = qkv + (size_t)(2 * HID) * NPIX;   // per-batch stride QKVO*NPIX

    // K1: qkv = W_qkv @ x          (M=1536, K=256)
    panel_gemm<CIN, false><<<dim3(NPIX / 64, QKVO / 128, BATCH), 256, 0, stream>>>(
        w_qkv, x, (size_t)CIN * NPIX, qkv, (size_t)QKVO * NPIX, nullptr);
    // K2: softmax over n on K part
    softmax_k<<<dim3(BATCH * HID), 256, 0, stream>>>(qkv);
    // K3: context = K_sm @ V^T
    context_gemm<<<dim3(8, 8, 32), 32, 0, stream>>>(qkv, ctx);
    // K4: attn = ctx^T @ Q  (into dead V region)
    attn_out_gemm<<<dim3(NPIX / 16, 8, 32), 32, 0, stream>>>(qkv, ctx);
    // K5: y = W_out @ attn + b     (M=256, K=512)
    panel_gemm<HID, true><<<dim3(NPIX / 64, CIN / 128, BATCH), 256, 0, stream>>>(
        w_out, attn, (size_t)QKVO * NPIX, y, (size_t)CIN * NPIX, b_out);
}
